// LinearAttention_83769042141682
// MI455X (gfx1250) — matrix operand order repaired and checked by reading, not compiled, not run on hardware
//
#include <hip/hip_runtime.h>
#include <hip/hip_bf16.h>
#include <math.h>

typedef __attribute__((ext_vector_type(16))) _Float16 v16h;
typedef __attribute__((ext_vector_type(8)))  _Float16 v8h;
typedef __attribute__((ext_vector_type(8)))  float    v8f;
typedef __attribute__((ext_vector_type(4)))  float    v4f;

union F16Frag { v16h v; v8h h[2]; };

static constexpr int BATCH = 4;
static constexpr int SEQ   = 4096;
static constexpr int DM    = 1024;
static constexpr int NH    = 16;
static constexpr int HD    = 64;
static constexpr int MTOT  = BATCH * SEQ;   // 16384

__device__ __forceinline__ v8f wmma_f16(v16h a, v16h b, v8f c) {
    // D = A(16x32 f16) * B(32x16 f16) + C(16x16 f32)
    return __builtin_amdgcn_wmma_f32_16x16x32_f16(
        false, a, false, b, (short)0, c, false, false);
}

// ---------------------------------------------------------------------------
// f32 -> f16 convert (vectorized, n divisible by 2048)
// ---------------------------------------------------------------------------
__global__ void cvt_kernel(const float* __restrict__ in, _Float16* __restrict__ out, int n) {
    int i = (blockIdx.x * blockDim.x + threadIdx.x) * 8;
    if (i + 8 <= n) {
        v4f a = *(const v4f*)(in + i);
        v4f b = *(const v4f*)(in + i + 4);
        v8h o;
        o[0] = (_Float16)a[0]; o[1] = (_Float16)a[1];
        o[2] = (_Float16)a[2]; o[3] = (_Float16)a[3];
        o[4] = (_Float16)b[0]; o[5] = (_Float16)b[1];
        o[6] = (_Float16)b[2]; o[7] = (_Float16)b[3];
        *(v8h*)(out + i) = o;
    }
}

// ---------------------------------------------------------------------------
// C[m,n] = epilogue( sum_k A[m,k]*W[n,k] + bias[n] )
// A: [M,K] f16 row-major.  W: [N,K] f16 row-major (torch Linear weight).
// MODE 0: elu1 -> f16   MODE 1: identity -> f16   MODE 2: identity -> f32
// Grid: (N/256, M/128), block 256 (8 waves, 2Mx4N), wave tile 64Mx64N, K-step 32.
// Per wave per K-step: 16 ds_load_b128 feed 16 WMMAs.
// ---------------------------------------------------------------------------
template <int MODE>
__global__ __launch_bounds__(256) void gemm_xwT(
    const _Float16* __restrict__ A, const _Float16* __restrict__ W,
    const float* __restrict__ bias, void* __restrict__ Cout,
    int M, int N, int K)
{
    // tiles: rows x 32 halfs, stored as 5 v8h per row (4 data + 1 pad)
    __shared__ v8h lA[2][128 * 5];   // 20 KB
    __shared__ v8h lW[2][256 * 5];   // 40 KB

    const int tid  = threadIdx.x;
    const int lane = tid & 31;
    const int wave = tid >> 5;
    const int half = lane >> 4;     // lane half selects K sub-block per ISA layout
    const int lr   = lane & 15;
    const int waveM = wave & 1;     // 0..1  -> 64 rows
    const int waveN = wave >> 1;    // 0..3  -> 64 cols
    const int mbase = blockIdx.y * 128;
    const int nbase = blockIdx.x * 256;

    // staging: thread t -> A row t>>1 (2 v8h), W rows t>>1 and 128+(t>>1) (4 v8h)
    const int srow = tid >> 1;
    const int skof = (tid & 1) * 2;   // v8h units: 0 or 2 (halfs 0 or 16)

    v8f acc[4][4];
    #pragma unroll
    for (int i = 0; i < 4; ++i)
        #pragma unroll
        for (int j = 0; j < 4; ++j) acc[i][j] = {};

    const int nk = K >> 5;

    { // stage k-step 0
        const _Float16* ga  = A + (size_t)(mbase + srow) * K + skof * 8;
        const _Float16* gw0 = W + (size_t)(nbase + srow) * K + skof * 8;
        const _Float16* gw1 = W + (size_t)(nbase + 128 + srow) * K + skof * 8;
        lA[0][srow * 5 + skof]             = *(const v8h*)ga;
        lA[0][srow * 5 + skof + 1]         = *(const v8h*)(ga + 8);
        lW[0][srow * 5 + skof]             = *(const v8h*)gw0;
        lW[0][srow * 5 + skof + 1]         = *(const v8h*)(gw0 + 8);
        lW[0][(128 + srow) * 5 + skof]     = *(const v8h*)gw1;
        lW[0][(128 + srow) * 5 + skof + 1] = *(const v8h*)(gw1 + 8);
    }
    __syncthreads();

    for (int ks = 0; ks < nk; ++ks) {
        const int cur = ks & 1;
        if (ks + 1 < nk) {
            const int nxt = cur ^ 1;
            const size_t ko = (size_t)(ks + 1) * 32 + skof * 8;
            const _Float16* ga  = A + (size_t)(mbase + srow) * K + ko;
            const _Float16* gw0 = W + (size_t)(nbase + srow) * K + ko;
            const _Float16* gw1 = W + (size_t)(nbase + 128 + srow) * K + ko;
            lA[nxt][srow * 5 + skof]             = *(const v8h*)ga;
            lA[nxt][srow * 5 + skof + 1]         = *(const v8h*)(ga + 8);
            lW[nxt][srow * 5 + skof]             = *(const v8h*)gw0;
            lW[nxt][srow * 5 + skof + 1]         = *(const v8h*)(gw0 + 8);
            lW[nxt][(128 + srow) * 5 + skof]     = *(const v8h*)gw1;
            lW[nxt][(128 + srow) * 5 + skof + 1] = *(const v8h*)(gw1 + 8);
        }
        F16Frag af[4];
        #pragma unroll
        for (int mf = 0; mf < 4; ++mf) {
            const int row = waveM * 64 + mf * 16 + lr;
            // A 16-bit layout: lanes 0-15 hold K 0..7 & 16..23, lanes 16-31 hold 8..15 & 24..31
            af[mf].h[0] = lA[cur][row * 5 + half];
            af[mf].h[1] = lA[cur][row * 5 + 2 + half];
        }
        #pragma unroll
        for (int nf = 0; nf < 4; ++nf) {
            const int row = waveN * 64 + nf * 16 + lr;
            // B 16-bit layout: lanes 0-15 hold K 0..7 & 16..23, lanes 16-31 hold 8..15 & 24..31
            F16Frag bf;
            bf.h[0] = lW[cur][row * 5 + half];
            bf.h[1] = lW[cur][row * 5 + 2 + half];
            #pragma unroll
            for (int mf = 0; mf < 4; ++mf)
                acc[mf][nf] = wmma_f16(af[mf].v, bf.v, acc[mf][nf]);
        }
        __syncthreads();
    }

    // epilogue: C layout VGPR r, lane L -> m = r + (L>=16 ? 8 : 0), n = L%16
    #pragma unroll
    for (int mf = 0; mf < 4; ++mf) {
        #pragma unroll
        for (int nf = 0; nf < 4; ++nf) {
            const int n = nbase + waveN * 64 + nf * 16 + lr;
            const float bv = bias[n];
            #pragma unroll
            for (int r = 0; r < 8; ++r) {
                const int m = mbase + waveM * 64 + mf * 16 + r + half * 8;
                float val = acc[mf][nf][r] + bv;
                if constexpr (MODE == 0) val = (val > 0.f) ? (val + 1.f) : __expf(val);
                if constexpr (MODE == 2)
                    ((float*)Cout)[(size_t)m * N + n] = val;
                else
                    ((_Float16*)Cout)[(size_t)m * N + n] = (_Float16)val;
            }
        }
    }
}

// ---------------------------------------------------------------------------
// KV partials: per (b,h) KV[d,e] = sum_s K[s,d]*V[s,e], S split across 16 blocks.
// Grid (16, 64). Block 256 s-rows, transposed LDS staging, 8 waves share tiles:
// wave w -> d-frag (w&3), e-frags (w>>2) and (w>>2)+2.
// KVpart: [16][bh][64][64] f32 (deterministic, no atomics)
// ---------------------------------------------------------------------------
__global__ __launch_bounds__(256) void kv_kernel(
    const _Float16* __restrict__ Kp, const _Float16* __restrict__ Vp,
    float* __restrict__ KVpart)
{
    __shared__ alignas(16) _Float16 lKt[64 * 40]; // [d][s], stride 40 halfs
    __shared__ alignas(16) _Float16 lVt[64 * 40]; // [e][s]

    const int tid  = threadIdx.x;
    const int lane = tid & 31;
    const int wave = tid >> 5;
    const int half = lane >> 4;
    const int lr   = lane & 15;
    const int bh   = blockIdx.y;
    const int b    = bh >> 4, h = bh & 15;
    const int s0blk = blockIdx.x * 256;

    const int mf  = wave & 3;
    const int nf0 = wave >> 2;

    v8f acc0 = {}, acc1 = {};

    const int sidx = tid >> 3;          // 0..31
    const int d0   = (tid & 7) * 8;     // 0..56

    for (int ks = 0; ks < 8; ++ks) {
        const int s = s0blk + ks * 32 + sidx;
        const size_t rowoff = (size_t)(b * SEQ + s) * DM + h * HD + d0;
        v8h kv = *(const v8h*)(Kp + rowoff);
        v8h vv = *(const v8h*)(Vp + rowoff);
        #pragma unroll
        for (int j = 0; j < 8; ++j) {     // transpose into LDS
            lKt[(d0 + j) * 40 + sidx] = kv[j];
            lVt[(d0 + j) * 40 + sidx] = vv[j];
        }
        __syncthreads();
        F16Frag af, bf0, bf1;
        {
            const int row = mf * 16 + lr;
            af.h[0] = *(const v8h*)&lKt[row * 40 + half * 8];
            af.h[1] = *(const v8h*)&lKt[row * 40 + 16 + half * 8];
        }
        {
            int row = nf0 * 16 + lr;
            bf0.h[0] = *(const v8h*)&lVt[row * 40 + half * 8];
            bf0.h[1] = *(const v8h*)&lVt[row * 40 + 16 + half * 8];
            row = (nf0 + 2) * 16 + lr;
            bf1.h[0] = *(const v8h*)&lVt[row * 40 + half * 8];
            bf1.h[1] = *(const v8h*)&lVt[row * 40 + 16 + half * 8];
        }
        acc0 = wmma_f16(af.v, bf0.v, acc0);
        acc1 = wmma_f16(af.v, bf1.v, acc1);
        __syncthreads();
    }

    float* dst = KVpart + ((size_t)blockIdx.x << 18) + ((size_t)bh << 12);
    #pragma unroll
    for (int r = 0; r < 8; ++r) {
        const int d = mf * 16 + r + half * 8;
        dst[d * 64 + nf0 * 16 + lr]       = acc0[r];
        dst[d * 64 + (nf0 + 2) * 16 + lr] = acc1[r];
    }
}

// Reduce 16 partials, write KV^T in f16: KVt[bh][e][d]
__global__ void kvred_kernel(const float* __restrict__ KVpart, _Float16* __restrict__ KVt) {
    const int i = blockIdx.x * 256 + threadIdx.x;   // 0 .. 64*4096-1
    const int bh = i >> 12, r = i & 4095, d = r >> 6, e = r & 63;
    float acc = 0.f;
    #pragma unroll
    for (int p = 0; p < 16; ++p) acc += KVpart[((size_t)p << 18) + i];
    KVt[((size_t)bh << 12) + e * 64 + d] = (_Float16)acc;
}

// Kspart[split][bh][d] = sum over 256 s-rows of K[b,s,h*64+d]  (grid 16x64)
__global__ __launch_bounds__(256) void ksum_kernel(
    const _Float16* __restrict__ Kp, float* __restrict__ Kspart)
{
    __shared__ float lsum[256];
    const int tid = threadIdx.x;
    const int bh = blockIdx.y, b = bh >> 4, h = bh & 15;
    const int s0 = blockIdx.x * 256;
    const int d = tid & 63, part = tid >> 6;
    float acc = 0.f;
    #pragma unroll 4
    for (int i = 0; i < 64; ++i) {
        const int s = s0 + part + i * 4;
        acc += (float)Kp[(size_t)(b * SEQ + s) * DM + h * HD + d];
    }
    lsum[tid] = acc;
    __syncthreads();
    if (part == 0)
        Kspart[(size_t)blockIdx.x * 4096 + bh * 64 + d] =
            lsum[d] + lsum[64 + d] + lsum[128 + d] + lsum[192 + d];
}

// rnorm[bh][s] = 1 / (Q[b,s,h].Ksum[bh] + 1e-6), Ksum folded from 16 partials
__global__ __launch_bounds__(256) void norm_kernel(
    const _Float16* __restrict__ Qp, const float* __restrict__ Kspart,
    float* __restrict__ rnorm)
{
    __shared__ float lks[64];
    const int bh = blockIdx.y, b = bh >> 4, h = bh & 15;
    const int s = blockIdx.x * 256 + threadIdx.x;
    if (threadIdx.x < 64) {
        float a = 0.f;
        #pragma unroll
        for (int p = 0; p < 16; ++p)
            a += Kspart[(size_t)p * 4096 + bh * 64 + threadIdx.x];
        lks[threadIdx.x] = a;
    }
    __syncthreads();
    const v8h* q = (const v8h*)(Qp + (size_t)(b * SEQ + s) * DM + h * HD);
    float acc = 0.f;
    #pragma unroll
    for (int i = 0; i < 8; ++i) {
        v8h x = q[i];
        #pragma unroll
        for (int j = 0; j < 8; ++j) acc += (float)x[j] * lks[i * 8 + j];
    }
    rnorm[bh * SEQ + s] = 1.f / (acc + 1e-6f);
}

// O[b,s,h*64+e] = rnorm * sum_d Q[b,s,h*64+d] * KV[d,e]   (B from KVt, contiguous)
// Grid (S/128, 64), 8 waves, wave w -> 16 rows, N=64, K=64 in 2 steps.
__global__ __launch_bounds__(256) void attn_kernel(
    const _Float16* __restrict__ Qp, const _Float16* __restrict__ KVt,
    const float* __restrict__ rnorm, _Float16* __restrict__ Op)
{
    const int tid = threadIdx.x, lane = tid & 31, wave = tid >> 5;
    const int half = lane >> 4, lr = lane & 15;
    const int bh = blockIdx.y, b = bh >> 4, h = bh & 15;
    const int srow = blockIdx.x * 128 + wave * 16 + lr;

    v8f acc[4];
    #pragma unroll
    for (int i = 0; i < 4; ++i) acc[i] = {};

    const _Float16* qrow = Qp + (size_t)(b * SEQ + srow) * DM + h * HD;
    const _Float16* kvb  = KVt + ((size_t)bh << 12);

    #pragma unroll
    for (int kk = 0; kk < 2; ++kk) {
        F16Frag a;
        a.h[0] = *(const v8h*)(qrow + kk * 32 + half * 8);
        a.h[1] = *(const v8h*)(qrow + kk * 32 + 16 + half * 8);
        #pragma unroll
        for (int nf = 0; nf < 4; ++nf) {
            F16Frag bfr;
            const _Float16* kr = kvb + (nf * 16 + lr) * 64 + kk * 32 + half * 8;
            bfr.h[0] = *(const v8h*)kr;
            bfr.h[1] = *(const v8h*)(kr + 16);
            acc[nf] = wmma_f16(a.v, bfr.v, acc[nf]);
        }
    }

    const int sbase = blockIdx.x * 128 + wave * 16;
    #pragma unroll
    for (int r = 0; r < 8; ++r) {
        const int m = sbase + r + half * 8;
        const float rn = rnorm[bh * SEQ + m];
        #pragma unroll
        for (int nf = 0; nf < 4; ++nf)
            Op[(size_t)(b * SEQ + m) * DM + h * HD + nf * 16 + lr] =
                (_Float16)(acc[nf][r] * rn);
    }
}

// ---------------------------------------------------------------------------
extern "C" void kernel_launch(void* const* d_in, const int* in_sizes, int n_in,
                              void* d_out, int out_size, void* d_ws, size_t ws_size,
                              hipStream_t stream) {
    const float* query = (const float*)d_in[0];
    const float* key   = (const float*)d_in[1];
    const float* value = (const float*)d_in[2];
    const float* Wq = (const float*)d_in[3];  const float* bq = (const float*)d_in[4];
    const float* Wk = (const float*)d_in[5];  const float* bk = (const float*)d_in[6];
    const float* Wv = (const float*)d_in[7];  const float* bv = (const float*)d_in[8];
    const float* Wo = (const float*)d_in[9];  const float* bo = (const float*)d_in[10];
    float* out = (float*)d_out;

    const size_t ACT = (size_t)MTOT * DM;           // 16,777,216 elements
    const size_t WEL = (size_t)DM * DM;             // 1,048,576 elements

    char* w = (char*)d_ws;
    _Float16* cq  = (_Float16*)w; w += ACT * 2;     // converted query (later reused as Oattn)
    _Float16* ck  = (_Float16*)w; w += ACT * 2;
    _Float16* cv  = (_Float16*)w; w += ACT * 2;
    _Float16* Q16 = (_Float16*)w; w += ACT * 2;
    _Float16* K16 = (_Float16*)w; w += ACT * 2;
    _Float16* V16 = (_Float16*)w; w += ACT * 2;
    _Float16* wq16 = (_Float16*)w; w += WEL * 2;
    _Float16* wk16 = (_Float16*)w; w += WEL * 2;
    _Float16* wv16 = (_Float16*)w; w += WEL * 2;
    _Float16* wo16 = (_Float16*)w; w += WEL * 2;
    float*    KVpart = (float*)w;  w += (size_t)16 * 64 * 4096 * 4;   // 16 MB
    _Float16* KVt  = (_Float16*)w; w += (size_t)64 * 4096 * 2;
    float*    Kspart = (float*)w;  w += (size_t)16 * 64 * 64 * 4;     // 256 KB
    float*    rnorm = (float*)w;   w += (size_t)64 * SEQ * 4;
    _Float16* Oattn = cq;  // reuse: cq consumed before attn_kernel writes

    // 1) convert activations + weights to f16
    cvt_kernel<<<ACT / 8 / 256, 256, 0, stream>>>(query, cq, (int)ACT);
    cvt_kernel<<<ACT / 8 / 256, 256, 0, stream>>>(key,   ck, (int)ACT);
    cvt_kernel<<<ACT / 8 / 256, 256, 0, stream>>>(value, cv, (int)ACT);
    cvt_kernel<<<WEL / 8 / 256, 256, 0, stream>>>(Wq, wq16, (int)WEL);
    cvt_kernel<<<WEL / 8 / 256, 256, 0, stream>>>(Wk, wk16, (int)WEL);
    cvt_kernel<<<WEL / 8 / 256, 256, 0, stream>>>(Wv, wv16, (int)WEL);
    cvt_kernel<<<WEL / 8 / 256, 256, 0, stream>>>(Wo, wo16, (int)WEL);

    // 2) projections (Q,K with fused elu1)
    dim3 ggrid(DM / 256, MTOT / 128);   // (4, 128)
    gemm_xwT<0><<<ggrid, 256, 0, stream>>>(cq, wq16, bq, Q16, MTOT, DM, DM);
    gemm_xwT<0><<<ggrid, 256, 0, stream>>>(ck, wk16, bk, K16, MTOT, DM, DM);
    gemm_xwT<1><<<ggrid, 256, 0, stream>>>(cv, wv16, bv, V16, MTOT, DM, DM);

    // 3) KV = K^T V per head (split-S partials, deterministic reduce -> KV^T f16)
    kv_kernel<<<dim3(16, 64), 256, 0, stream>>>(K16, V16, KVpart);
    kvred_kernel<<<(64 * 4096) / 256, 256, 0, stream>>>(KVpart, KVt);

    // 4) normalizer (split-S K column sums, folded in norm_kernel)
    ksum_kernel<<<dim3(16, 64), 256, 0, stream>>>(K16, Kspart);
    norm_kernel<<<dim3(SEQ / 256, 64), 256, 0, stream>>>(Q16, Kspart, rnorm);

    // 5) O = (Q @ KV) * rnorm
    attn_kernel<<<dim3(SEQ / 128, 64), 256, 0, stream>>>(Q16, KVt, rnorm, Oattn);

    // 6) final projection -> f32 output
    gemm_xwT<2><<<ggrid, 256, 0, stream>>>(Oattn, wo16, bo, (void*)out, MTOT, DM, DM);
}
